// GNSMsg_EdgeSelfAttnKHop_81612968558665
// MI455X (gfx1250) — matrix operand order repaired and checked by reading, not compiled, not run on hardware
//
#include <hip/hip_runtime.h>
#include <hip/hip_bf16.h>
#include <math.h>

// ---------------- problem constants ----------------
#define Bn   4
#define Nn   2048
#define En   16384
#define Dn   10
#define DMn  256
#define NHn  8
#define DHn  32
#define FEn  9
#define Kn   30
#define BN   (Bn * Nn)          // 8192
#define ETOT (Bn * En)          // 65536
#define NKT  (Nn / 32)          // 64 k-tiles per batch row-block for k-hop

// hop weights  exp(-h^2 / (2*1.5^2)), h=1..3
#define HOPW1 0.8007374029168081f
#define HOPW2 0.4111122905071874f
#define HOPW3 0.1353352832366127f
#define HOPZ  1.3471849766606082f
#define INV_SQRT_DH 0.17677669529663687f

typedef __attribute__((ext_vector_type(16))) _Float16 v16h;
typedef __attribute__((ext_vector_type(8)))  float    v8f;

union AV { v16h v; uint4 u[2]; };

__device__ __forceinline__ float waveSum(float v) {
    #pragma unroll
    for (int o = 16; o > 0; o >>= 1) v += __shfl_xor(v, o, 32);
    return v;
}
__device__ __forceinline__ unsigned fenc(float x) {
    unsigned u = __float_as_uint(x);
    return (u & 0x80000000u) ? ~u : (u | 0x80000000u);
}
__device__ __forceinline__ float fdec(unsigned k) {
    return __uint_as_float((k & 0x80000000u) ? (k ^ 0x80000000u) : ~k);
}

// ---------------- init: state from V0, m=0 ----------------
__global__ void k_init_state(const float* __restrict__ V0, float* __restrict__ vst,
                             float* __restrict__ thst, float* __restrict__ mst) {
    int i = blockIdx.x * blockDim.x + threadIdx.x;
    if (i >= BN) return;
    vst[i]  = V0[2 * i];
    thst[i] = V0[2 * i + 1];
    #pragma unroll
    for (int d = 0; d < Dn; ++d) mst[i * Dn + d] = 0.0f;
}

// ---------------- build T (row-normalized |Y|) as f16 ----------------
__global__ void k_buildT(const float* __restrict__ G, const float* __restrict__ Bm,
                         _Float16* __restrict__ Th) {
    int row = blockIdx.x;            // global row in [0, B*N)
    int b = row >> 11, i = row & (Nn - 1);
    size_t base = (size_t)b * Nn * Nn + (size_t)i * Nn;
    int tid = threadIdx.x;
    __shared__ float sd[256];
    float s = 0.0f;
    for (int j = tid; j < Nn; j += 256) {
        float g = G[base + j], bm = Bm[base + j];
        float a = sqrtf(g * g + bm * bm) * (j == i ? 0.0f : 1.0f);
        s += a;
    }
    sd[tid] = s; __syncthreads();
    for (int st = 128; st > 0; st >>= 1) { if (tid < st) sd[tid] += sd[tid + st]; __syncthreads(); }
    float inv = 1.0f / fmaxf(sd[0], 1e-12f);
    __syncthreads();
    for (int j = tid; j < Nn; j += 256) {
        float g = G[base + j], bm = Bm[base + j];
        float a = sqrtf(g * g + bm * bm) * (j == i ? 0.0f : 1.0f);
        Th[base + j] = (_Float16)(a * inv);
    }
}

// ---------------- pack a K x Nc f32 weight into WMMA B-fragment layout ----------------
// out[((kt*Ntiles+nt)*32 + lane)*16 + t] = W[kt*32 + kb + t + (t<8?0:8), nt*16 + (lane&15)], kb=(lane<16?0:8)
__global__ void k_packW(const float* __restrict__ W, _Float16* __restrict__ out,
                        int Kdim, int Nc, int Ntiles) {
    int nt = blockIdx.x % Ntiles, kt = blockIdx.x / Ntiles;
    int L = threadIdx.x;
    int col = nt * 16 + (L & 15);
    int kb  = kt * 32 + ((L < 16) ? 0 : 8);
    size_t o = ((size_t)(kt * Ntiles + nt) * 32 + L) * 16;
    #pragma unroll
    for (int t = 0; t < 16; ++t) {
        int kk = kb + t + ((t < 8) ? 0 : 8);
        float v = (kk < Kdim && col < Nc) ? W[(size_t)kk * Nc + col] : 0.0f;
        out[o + t] = (_Float16)v;
    }
}

// ---------------- pack BN x 16 f16 row-major into per-batch B-fragment layout ----------------
__global__ void k_packHop(const _Float16* __restrict__ in, _Float16* __restrict__ out) {
    int blk = blockIdx.x;            // b*NKT + kt
    int b = blk / NKT, kt = blk % NKT;
    int L = threadIdx.x;
    int col = L & 15;
    int kb  = kt * 32 + ((L < 16) ? 0 : 8);
    size_t o = ((size_t)blk * 32 + L) * 16;
    #pragma unroll
    for (int t = 0; t < 16; ++t)
        out[o + t] = in[((size_t)b * Nn + kb + t + ((t < 8) ? 0 : 8)) * 16 + col];
}

// ---------------- e = v cos(th), f = v sin(th) ----------------
__global__ void k_ef(const float* __restrict__ vst, const float* __restrict__ thst,
                     float* __restrict__ ev, float* __restrict__ fv) {
    int i = blockIdx.x * blockDim.x + threadIdx.x;
    if (i >= BN) return;
    float v = vst[i], t = thst[i];
    ev[i] = v * __cosf(t);
    fv[i] = v * __sinf(t);
}

// ---------------- fused power-flow row pass + ns assembly + acc zero ----------------
__global__ void k_mismatch_ns(const float* __restrict__ G, const float* __restrict__ Bm,
                              const float* __restrict__ ev, const float* __restrict__ fv,
                              const int* __restrict__ bus_type,
                              const float* __restrict__ P_set, const float* __restrict__ Q_set,
                              const float* __restrict__ vst, const float* __restrict__ thst,
                              const float* __restrict__ mst,
                              float* __restrict__ ns, _Float16* __restrict__ hop16,
                              float* __restrict__ acc) {
    int row = blockIdx.x;
    int b = row >> 11, i = row & (Nn - 1);
    size_t base = (size_t)b * Nn * Nn + (size_t)i * Nn;
    const float* eb = ev + b * Nn;
    const float* fb = fv + b * Nn;
    int tid = threadIdx.x;
    float ge = 0.f, gf = 0.f, be = 0.f, bf = 0.f;
    for (int j = tid; j < Nn; j += 256) {
        float g = G[base + j], bm = Bm[base + j];
        float e = eb[j], f = fb[j];
        ge += g * e;  gf += g * f;  be += bm * e;  bf += bm * f;
    }
    __shared__ float sd[256];
    __shared__ float res[2];
    auto bred = [&](float v) -> float {
        sd[tid] = v; __syncthreads();
        for (int st = 128; st > 0; st >>= 1) { if (tid < st) sd[tid] += sd[tid + st]; __syncthreads(); }
        float r = sd[0]; __syncthreads(); return r;
    };
    ge = bred(ge); gf = bred(gf); be = bred(be); bf = bred(bf);
    if (tid == 0) {
        float Ire = ge - bf, Iim = gf + be;
        float e_i = ev[row], f_i = fv[row];
        int bt = bus_type[row];
        float pm = (bt == 1) ? 0.f : 1.f;
        float qm = (bt == 1 || bt == 2) ? 0.f : 1.f;
        res[0] = (P_set[row] - (e_i * Ire + f_i * Iim)) * pm;   // DP
        res[1] = (Q_set[row] - (f_i * Ire - e_i * Iim)) * qm;   // DQ
    }
    __syncthreads();
    if (tid < 16) {
        float val;
        if      (tid == 0) val = vst[row];
        else if (tid == 1) val = thst[row];
        else if (tid == 2) val = res[0];
        else if (tid == 3) val = res[1];
        else if (tid < 14) val = mst[row * Dn + (tid - 4)];
        else               val = 0.0f;
        ns[row * 16 + tid]    = val;
        hop16[row * 16 + tid] = (_Float16)val;
        acc[row * 16 + tid]   = 0.0f;
    }
}

// ---------------- one k-hop: H_new = T @ H (WMMA), 64x16 per wave ----------------
// B fragment is shared across all row tiles -> 4 M-tiles per wave, B loaded once per k-step.
__global__ void k_hop(const _Float16* __restrict__ Th, const _Float16* __restrict__ Bp,
                      _Float16* __restrict__ hop16, float* __restrict__ acc, float w) {
    int gw = blockIdx.x * 4 + (threadIdx.x >> 5);   // global wave
    int lane = threadIdx.x & 31;
    if (gw >= Bn * (Nn / 64)) return;
    int b = gw / (Nn / 64), mtg = gw % (Nn / 64);
    int kb = (lane < 16) ? 0 : 8;
    const _Float16* Arow[4];
    #pragma unroll
    for (int i = 0; i < 4; ++i) {
        int m = (mtg * 4 + i) * 16 + (lane & 15);
        Arow[i] = Th + (size_t)b * Nn * Nn + (size_t)m * Nn;
    }
    v8f c0 = {}, c1 = {}, c2 = {}, c3 = {};
    for (int kt = 0; kt < NKT; ++kt) {
        AV bb;
        const uint4* bp = (const uint4*)(Bp + ((size_t)(b * NKT + kt) * 32 + lane) * 16);
        bb.u[0] = bp[0]; bb.u[1] = bp[1];
        AV a0, a1, a2, a3;
        a0.u[0] = *(const uint4*)(Arow[0] + kt * 32 + kb);
        a0.u[1] = *(const uint4*)(Arow[0] + kt * 32 + kb + 16);
        a1.u[0] = *(const uint4*)(Arow[1] + kt * 32 + kb);
        a1.u[1] = *(const uint4*)(Arow[1] + kt * 32 + kb + 16);
        a2.u[0] = *(const uint4*)(Arow[2] + kt * 32 + kb);
        a2.u[1] = *(const uint4*)(Arow[2] + kt * 32 + kb + 16);
        a3.u[0] = *(const uint4*)(Arow[3] + kt * 32 + kb);
        a3.u[1] = *(const uint4*)(Arow[3] + kt * 32 + kb + 16);
        if (kt + 1 < NKT) {
            __builtin_prefetch(Arow[0] + (kt + 1) * 32 + kb, 0, 3);
            __builtin_prefetch(Arow[2] + (kt + 1) * 32 + kb, 0, 3);
        }
        c0 = __builtin_amdgcn_wmma_f32_16x16x32_f16(false, a0.v, false, bb.v, (short)0, c0, false, false);
        c1 = __builtin_amdgcn_wmma_f32_16x16x32_f16(false, a1.v, false, bb.v, (short)0, c1, false, false);
        c2 = __builtin_amdgcn_wmma_f32_16x16x32_f16(false, a2.v, false, bb.v, (short)0, c2, false, false);
        c3 = __builtin_amdgcn_wmma_f32_16x16x32_f16(false, a3.v, false, bb.v, (short)0, c3, false, false);
    }
    int col = lane & 15;
    #pragma unroll
    for (int i = 0; i < 4; ++i) {
        v8f c = (i == 0) ? c0 : (i == 1) ? c1 : (i == 2) ? c2 : c3;
        int rbase = (mtg * 4 + i) * 16 + ((lane < 16) ? 0 : 8);
        #pragma unroll
        for (int r = 0; r < 8; ++r) {
            int row = rbase + r;
            size_t o = ((size_t)b * Nn + row) * 16 + col;
            float v = c[r];
            hop16[o] = (_Float16)v;
            acc[o]  += w * v;
        }
    }
}

// ---------------- xin = [ns(14) | khop/Z(14) | 0,0,0,0] as f16 ----------------
__global__ void k_build_xin(const float* __restrict__ ns, const float* __restrict__ acc,
                            _Float16* __restrict__ xin) {
    int i = blockIdx.x * blockDim.x + threadIdx.x;
    if (i >= BN * 32) return;
    int row = i >> 5, c = i & 31;
    float v = 0.0f;
    if (c < 14)      v = ns[row * 16 + c];
    else if (c < 28) v = acc[row * 16 + (c - 14)] * (1.0f / HOPZ);
    xin[i] = (_Float16)v;
}

// ---------------- WMMA GEMM, 16x64 strip per wave (4 N-tiles, A reused) ----------------
// C = A(f16,MxK) @ Bp + bias (+resid)(+relu); Ntiles must be a multiple of 4.
__global__ void k_gemm(const _Float16* __restrict__ A, const _Float16* __restrict__ Bp,
                       const float* __restrict__ bias, const float* __restrict__ resid,
                       float* __restrict__ C, _Float16* __restrict__ Ch,
                       int M, int Kt, int Ntiles, int relu) {
    int gw = blockIdx.x * 4 + (threadIdx.x >> 5);
    int lane = threadIdx.x & 31;
    int ngrp = Ntiles >> 2;
    int strips = (M / 16) * ngrp;
    if (gw >= strips) return;
    int mt = gw / ngrp, ng = gw % ngrp;
    int Kdim = Kt * 32, Nc = Ntiles * 16;
    int m  = mt * 16 + (lane & 15);
    int kb = (lane < 16) ? 0 : 8;
    const _Float16* Arow = A + (size_t)m * Kdim;
    v8f c0 = {}, c1 = {}, c2 = {}, c3 = {};
    for (int kt = 0; kt < Kt; ++kt) {
        AV a;
        a.u[0] = *(const uint4*)(Arow + kt * 32 + kb);
        a.u[1] = *(const uint4*)(Arow + kt * 32 + kb + 16);
        const uint4* bp = (const uint4*)(Bp + ((size_t)(kt * Ntiles + ng * 4) * 32 + lane) * 16);
        AV b0, b1, b2, b3;
        b0.u[0] = bp[0];       b0.u[1] = bp[1];
        b1.u[0] = bp[0 + 64];  b1.u[1] = bp[1 + 64];   // next n-tile: +32 lanes * 16 f16 = +64 uint4
        b2.u[0] = bp[0 + 128]; b2.u[1] = bp[1 + 128];
        b3.u[0] = bp[0 + 192]; b3.u[1] = bp[1 + 192];
        if (kt + 1 < Kt) {
            __builtin_prefetch(Arow + (kt + 1) * 32 + kb, 0, 3);
            __builtin_prefetch(Bp + ((size_t)((kt + 1) * Ntiles + ng * 4) * 32 + lane) * 16, 0, 3);
        }
        c0 = __builtin_amdgcn_wmma_f32_16x16x32_f16(false, a.v, false, b0.v, (short)0, c0, false, false);
        c1 = __builtin_amdgcn_wmma_f32_16x16x32_f16(false, a.v, false, b1.v, (short)0, c1, false, false);
        c2 = __builtin_amdgcn_wmma_f32_16x16x32_f16(false, a.v, false, b2.v, (short)0, c2, false, false);
        c3 = __builtin_amdgcn_wmma_f32_16x16x32_f16(false, a.v, false, b3.v, (short)0, c3, false, false);
    }
    int rbase = mt * 16 + ((lane < 16) ? 0 : 8);
    #pragma unroll
    for (int i = 0; i < 4; ++i) {
        v8f c = (i == 0) ? c0 : (i == 1) ? c1 : (i == 2) ? c2 : c3;
        int col = (ng * 4 + i) * 16 + (lane & 15);
        float bv = bias ? bias[col] : 0.0f;
        #pragma unroll
        for (int r = 0; r < 8; ++r) {
            int row = rbase + r;
            float v = c[r] + bv;
            if (resid) v += resid[(size_t)row * Nc + col];
            if (relu)  v = fmaxf(v, 0.0f);
            C[(size_t)row * Nc + col] = v;
            if (Ch) Ch[(size_t)row * Nc + col] = (_Float16)v;
        }
    }
}

// ---------------- LayerNorm (256 cols), one wave per row ----------------
__global__ void k_ln(const float* __restrict__ in, const float* __restrict__ g,
                     const float* __restrict__ bta, float* __restrict__ out,
                     _Float16* __restrict__ outh) {
    int row = blockIdx.x * 4 + (threadIdx.x >> 5);
    int lane = threadIdx.x & 31;
    if (row >= BN) return;
    const float* p = in + (size_t)row * DMn;
    float x[8]; float s = 0.f, s2 = 0.f;
    #pragma unroll
    for (int i = 0; i < 8; ++i) { x[i] = p[lane + 32 * i]; s += x[i]; s2 += x[i] * x[i]; }
    s = waveSum(s); s2 = waveSum(s2);
    float mu = s * (1.0f / DMn);
    float var = s2 * (1.0f / DMn) - mu * mu;
    float rs = rsqrtf(var + 1e-5f);
    #pragma unroll
    for (int i = 0; i < 8; ++i) {
        int c = lane + 32 * i;
        float v = g[c] * (x[i] - mu) * rs + bta[c];
        out[(size_t)row * DMn + c]  = v;
        outh[(size_t)row * DMn + c] = (_Float16)v;
    }
}

// ---------------- edge bias eb = edge_feat @ We + be (once) ----------------
__global__ void k_eb(const float* __restrict__ ef, const float* __restrict__ We,
                     const float* __restrict__ be, float* __restrict__ eb) {
    int e = blockIdx.x * blockDim.x + threadIdx.x;
    if (e >= ETOT) return;
    float f[FEn];
    #pragma unroll
    for (int i = 0; i < FEn; ++i) f[i] = ef[e * FEn + i];
    #pragma unroll
    for (int h = 0; h < NHn; ++h) {
        float s = be[h];
        #pragma unroll
        for (int i = 0; i < FEn; ++i) s += f[i] * We[i * NHn + h];
        eb[e * NHn + h] = s;
    }
}

// ---------------- attn scratch init ----------------
__global__ void k_attn_init(unsigned* __restrict__ mx, float* __restrict__ den,
                            float* __restrict__ agg) {
    int i = blockIdx.x * blockDim.x + threadIdx.x;
    if (i < BN * DMn) agg[i] = 0.0f;
    if (i < BN * NHn) { mx[i] = 0u; den[i] = 0.0f; }
}

// ---------------- edge scores + segment max ----------------
__global__ void k_score(const int* __restrict__ ei, const float* __restrict__ q,
                        const float* __restrict__ k, const float* __restrict__ eb,
                        float* __restrict__ sc, unsigned* __restrict__ mx) {
    int gid = blockIdx.x * blockDim.x + threadIdx.x;
    if (gid >= ETOT * NHn) return;
    int eg = gid >> 3, h = gid & 7;
    int b = eg / En, e = eg % En;
    int src = ei[b * 2 * En + e] + b * Nn;
    int dst = ei[b * 2 * En + En + e] + b * Nn;
    const float* qp = q + (size_t)dst * DMn + h * DHn;
    const float* kp = k + (size_t)src * DMn + h * DHn;
    float s = 0.f;
    #pragma unroll
    for (int d = 0; d < DHn; ++d) s += qp[d] * kp[d];
    s = s * INV_SQRT_DH + eb[eg * NHn + h];
    sc[gid] = s;
    atomicMax(&mx[dst * NHn + h], fenc(s));
}

// ---------------- p = exp(score - mx), segment sum ----------------
__global__ void k_pden(const int* __restrict__ ei, float* __restrict__ sc,
                       const unsigned* __restrict__ mx, float* __restrict__ den) {
    int gid = blockIdx.x * blockDim.x + threadIdx.x;
    if (gid >= ETOT * NHn) return;
    int eg = gid >> 3, h = gid & 7;
    int b = eg / En, e = eg % En;
    int dst = ei[b * 2 * En + En + e] + b * Nn;
    float p = __expf(sc[gid] - fdec(mx[dst * NHn + h]));
    sc[gid] = p;
    atomicAdd(&den[dst * NHn + h], p);
}

// ---------------- aggregate alpha * v into agg, one wave per edge ----------------
__global__ void k_agg(const int* __restrict__ ei, const float* __restrict__ p,
                      const float* __restrict__ den, const float* __restrict__ v,
                      float* __restrict__ agg) {
    int eg = blockIdx.x * 4 + (threadIdx.x >> 5);
    int lane = threadIdx.x & 31;
    if (eg >= ETOT) return;
    int b = eg / En, e = eg % En;
    int src = ei[b * 2 * En + e] + b * Nn;
    int dst = ei[b * 2 * En + En + e] + b * Nn;
    #pragma unroll
    for (int h = 0; h < NHn; ++h) {
        float a = p[eg * NHn + h] / den[dst * NHn + h];
        int c = h * DHn + lane;
        atomicAdd(&agg[(size_t)dst * DMn + c], a * v[(size_t)src * DMn + c]);
    }
}

// ---------------- f32 -> f16 convert ----------------
__global__ void k_cvt(const float* __restrict__ in, _Float16* __restrict__ out, int n) {
    int i = blockIdx.x * blockDim.x + threadIdx.x;
    if (i < n) out[i] = (_Float16)in[i];
}

// ---------------- per-step heads + state update, one wave per row ----------------
__global__ void k_heads(const float* __restrict__ x, const float* __restrict__ Wth,
                        const float* __restrict__ bth, const float* __restrict__ Wvh,
                        const float* __restrict__ bvh, const float* __restrict__ Wm,
                        const float* __restrict__ bm, const int* __restrict__ bus_type,
                        float* __restrict__ vst, float* __restrict__ thst,
                        float* __restrict__ mst) {
    int row = blockIdx.x * 4 + (threadIdx.x >> 5);
    int lane = threadIdx.x & 31;
    if (row >= BN) return;
    float s[12];
    #pragma unroll
    for (int i = 0; i < 12; ++i) s[i] = 0.0f;
    const float* xp = x + (size_t)row * DMn;
    for (int c = lane; c < DMn; c += 32) {
        float xv = xp[c];
        s[0] += xv * Wth[c];
        s[1] += xv * Wvh[c];
        #pragma unroll
        for (int d = 0; d < Dn; ++d) s[2 + d] += xv * Wm[c * Dn + d];
    }
    #pragma unroll
    for (int i = 0; i < 12; ++i) s[i] = waveSum(s[i]);
    if (lane == 0) {
        int bt = bus_type[row];
        float pm = (bt == 1) ? 0.f : 1.f;
        float qm = (bt == 1 || bt == 2) ? 0.f : 1.f;
        thst[row] += (s[0] + bth[0]) * pm;
        vst[row]  += (s[1] + bvh[0]) * qm;
        #pragma unroll
        for (int d = 0; d < Dn; ++d) mst[row * Dn + d] = s[2 + d] + bm[d];
    }
}

// ---------------- output pack ----------------
__global__ void k_out(const float* __restrict__ vst, const float* __restrict__ thst,
                      float* __restrict__ out) {
    int i = blockIdx.x * blockDim.x + threadIdx.x;
    if (i >= BN) return;
    out[2 * i]     = vst[i];
    out[2 * i + 1] = thst[i];
}

// ================== launch ==================
extern "C" void kernel_launch(void* const* d_in, const int* in_sizes, int n_in,
                              void* d_out, int out_size, void* d_ws, size_t ws_size,
                              hipStream_t stream) {
    const int*   bus_type = (const int*)  d_in[0];
    const int*   edge_idx = (const int*)  d_in[1];
    const float* G        = (const float*)d_in[2];
    const float* Bm       = (const float*)d_in[3];
    const float* P_set    = (const float*)d_in[4];
    const float* Q_set    = (const float*)d_in[5];
    const float* V0       = (const float*)d_in[6];
    const float* edgef    = (const float*)d_in[7];
    const float* Win      = (const float*)d_in[8];
    const float* b_in     = (const float*)d_in[9];
    const float* Wq = (const float*)d_in[10]; const float* bq = (const float*)d_in[11];
    const float* Wk = (const float*)d_in[12]; const float* bk = (const float*)d_in[13];
    const float* Wv = (const float*)d_in[14]; const float* bv = (const float*)d_in[15];
    const float* Wo = (const float*)d_in[16]; const float* bo = (const float*)d_in[17];
    const float* We = (const float*)d_in[18]; const float* be = (const float*)d_in[19];
    const float* ln1g = (const float*)d_in[20]; const float* ln1b = (const float*)d_in[21];
    const float* ln2g = (const float*)d_in[22]; const float* ln2b = (const float*)d_in[23];
    const float* Wf1 = (const float*)d_in[24]; const float* bf1 = (const float*)d_in[25];
    const float* Wf2 = (const float*)d_in[26]; const float* bf2 = (const float*)d_in[27];
    const float* Wth = (const float*)d_in[28]; const float* bth = (const float*)d_in[29];
    const float* Wvh = (const float*)d_in[30]; const float* bvh = (const float*)d_in[31];
    const float* Wm  = (const float*)d_in[32]; const float* bmH = (const float*)d_in[33];
    float* out = (float*)d_out;

    // -------- workspace carve --------
    size_t off = 0;
    auto carve = [&](size_t bytes) -> char* {
        char* p = (char*)d_ws + off;
        off = (off + bytes + 255) & ~(size_t)255;
        return p;
    };
    _Float16* Th     = (_Float16*)carve((size_t)Bn * Nn * Nn * 2);
    float*    ns     = (float*)   carve((size_t)BN * 16 * 4);
    float*    acc    = (float*)   carve((size_t)BN * 16 * 4);
    _Float16* hop16  = (_Float16*)carve((size_t)BN * 16 * 2);
    _Float16* hoppk  = (_Float16*)carve((size_t)BN * 16 * 2);
    _Float16* xin16  = (_Float16*)carve((size_t)BN * 32 * 2);
    float*    x      = (float*)   carve((size_t)BN * DMn * 4);
    _Float16* xh     = (_Float16*)carve((size_t)BN * DMn * 2);
    float*    qb     = (float*)   carve((size_t)BN * DMn * 4);
    float*    kb2    = (float*)   carve((size_t)BN * DMn * 4);
    float*    vb     = (float*)   carve((size_t)BN * DMn * 4);
    float*    tmp    = (float*)   carve((size_t)BN * DMn * 4);
    float*    agg    = (float*)   carve((size_t)BN * DMn * 4);
    _Float16* aggh   = (_Float16*)carve((size_t)BN * DMn * 2);
    float*    hbuf   = (float*)   carve((size_t)BN * 1024 * 4);
    _Float16* hh     = (_Float16*)carve((size_t)BN * 1024 * 2);
    float*    eb     = (float*)   carve((size_t)ETOT * NHn * 4);
    float*    sc     = (float*)   carve((size_t)ETOT * NHn * 4);
    unsigned* mx     = (unsigned*)carve((size_t)BN * NHn * 4);
    float*    den    = (float*)   carve((size_t)BN * NHn * 4);
    float*    vst    = (float*)   carve((size_t)BN * 4);
    float*    thst   = (float*)   carve((size_t)BN * 4);
    float*    mst    = (float*)   carve((size_t)BN * Dn * 4);
    float*    ev     = (float*)   carve((size_t)BN * 4);
    float*    fv     = (float*)   carve((size_t)BN * 4);
    _Float16* Winp   = (_Float16*)carve((size_t)1  * 16 * 512 * 2);
    _Float16* Wqp    = (_Float16*)carve((size_t)8  * 16 * 512 * 2);
    _Float16* Wkp    = (_Float16*)carve((size_t)8  * 16 * 512 * 2);
    _Float16* Wvp    = (_Float16*)carve((size_t)8  * 16 * 512 * 2);
    _Float16* Wop    = (_Float16*)carve((size_t)8  * 16 * 512 * 2);
    _Float16* Wf1p   = (_Float16*)carve((size_t)8  * 64 * 512 * 2);
    _Float16* Wf2p   = (_Float16*)carve((size_t)32 * 16 * 512 * 2);

    // -------- one-time prep --------
    k_init_state<<<(BN + 255) / 256, 256, 0, stream>>>(V0, vst, thst, mst);
    k_buildT<<<BN, 256, 0, stream>>>(G, Bm, Th);
    k_packW<<<1 * 16, 32, 0, stream>>>(Win, Winp, 28, DMn, 16);
    k_packW<<<8 * 16, 32, 0, stream>>>(Wq, Wqp, DMn, DMn, 16);
    k_packW<<<8 * 16, 32, 0, stream>>>(Wk, Wkp, DMn, DMn, 16);
    k_packW<<<8 * 16, 32, 0, stream>>>(Wv, Wvp, DMn, DMn, 16);
    k_packW<<<8 * 16, 32, 0, stream>>>(Wo, Wop, DMn, DMn, 16);
    k_packW<<<8 * 64, 32, 0, stream>>>(Wf1, Wf1p, DMn, 1024, 64);
    k_packW<<<32 * 16, 32, 0, stream>>>(Wf2, Wf2p, 1024, DMn, 16);
    k_eb<<<(ETOT + 255) / 256, 256, 0, stream>>>(edgef, We, be, eb);

    const float hw[3] = {HOPW1, HOPW2, HOPW3};
    // strips = (M/16) * (Ntiles/4); 4 waves per 128-thread block
    const int gemm256 = ((BN / 16) * (16 / 4)) / 4;   // Nc=256 -> 512 blocks
    const int gemm1k  = ((BN / 16) * (64 / 4)) / 4;   // Nc=1024 -> 2048 blocks
    const int hopblks = (Bn * (Nn / 64)) / 4;         // 32 blocks

    for (int step = 0; step < Kn; ++step) {
        // power-flow mismatch + ns
        k_ef<<<(BN + 255) / 256, 256, 0, stream>>>(vst, thst, ev, fv);
        k_mismatch_ns<<<BN, 256, 0, stream>>>(G, Bm, ev, fv, bus_type, P_set, Q_set,
                                              vst, thst, mst, ns, hop16, acc);
        // 3-hop propagation via WMMA
        for (int hcount = 0; hcount < 3; ++hcount) {
            k_packHop<<<Bn * NKT, 32, 0, stream>>>(hop16, hoppk);
            k_hop<<<hopblks, 128, 0, stream>>>(Th, hoppk, hop16, acc, hw[hcount]);
        }
        // input projection
        k_build_xin<<<(BN * 32 + 255) / 256, 256, 0, stream>>>(ns, acc, xin16);
        k_gemm<<<gemm256, 128, 0, stream>>>(xin16, Winp, b_in, nullptr, x, xh,
                                            BN, 1, 16, 0);
        // attention
        k_attn_init<<<(BN * DMn + 255) / 256, 256, 0, stream>>>(mx, den, agg);
        k_gemm<<<gemm256, 128, 0, stream>>>(xh, Wqp, bq, nullptr, qb, nullptr, BN, 8, 16, 0);
        k_gemm<<<gemm256, 128, 0, stream>>>(xh, Wkp, bk, nullptr, kb2, nullptr, BN, 8, 16, 0);
        k_gemm<<<gemm256, 128, 0, stream>>>(xh, Wvp, bv, nullptr, vb, nullptr, BN, 8, 16, 0);
        k_score<<<(ETOT * NHn + 255) / 256, 256, 0, stream>>>(edge_idx, qb, kb2, eb, sc, mx);
        k_pden<<<(ETOT * NHn + 255) / 256, 256, 0, stream>>>(edge_idx, sc, mx, den);
        k_agg<<<ETOT / 4, 128, 0, stream>>>(edge_idx, sc, den, vb, agg);
        k_cvt<<<(BN * DMn + 255) / 256, 256, 0, stream>>>(agg, aggh, BN * DMn);
        k_gemm<<<gemm256, 128, 0, stream>>>(aggh, Wop, bo, x, tmp, nullptr, BN, 8, 16, 0);
        k_ln<<<BN / 4, 128, 0, stream>>>(tmp, ln1g, ln1b, x, xh);
        // FFN
        k_gemm<<<gemm1k, 128, 0, stream>>>(xh, Wf1p, bf1, nullptr, hbuf, hh, BN, 8, 64, 1);
        k_gemm<<<gemm256, 128, 0, stream>>>(hh, Wf2p, bf2, x, tmp, nullptr, BN, 32, 16, 0);
        k_ln<<<BN / 4, 128, 0, stream>>>(tmp, ln2g, ln2b, x, xh);
        // per-step heads + state update
        k_heads<<<BN / 4, 128, 0, stream>>>(x, Wth + step * DMn, bth + step,
                                            Wvh + step * DMn, bvh + step,
                                            Wm + step * DMn * Dn, bmH + step * Dn,
                                            bus_type, vst, thst, mst);
    }
    k_out<<<(BN + 255) / 256, 256, 0, stream>>>(vst, thst, out);
}
